// PLE_12816182411988
// MI455X (gfx1250) — matrix operand order repaired and checked by reading, not compiled, not run on hardware
//
#include <hip/hip_runtime.h>
#include <hip/hip_bf16.h>

typedef __attribute__((ext_vector_type(16))) __bf16 v16bf;
typedef __attribute__((ext_vector_type(8)))  float  v8f;
typedef __attribute__((ext_vector_type(4)))  float  v4f;

// Problem constants (fixed by the reference)
constexpr int BATCH = 65536;
constexpr int DK    = 1024;   // reduction dim
constexpr int NCOL  = 352;    // 340 real cols padded to 22 tiles of 16
constexpr int NT    = 22;     // live N tiles
constexpr int NTP   = 24;     // padded N tiles (uniform async DMA: 6 b128/thread)
constexpr int NTW   = 11;     // live N tiles per wave (N split across wave pairs)
constexpr int KC    = 32;     // K chunks of 32
constexpr int CHUNK_DW    = NTP * 32 * 8;        // 6144 dwords per K-chunk of packed B
constexpr int CHUNK_BYTES = CHUNK_DW * 4;        // 24576
constexpr int BPACK_DW    = KC * CHUNK_DW;       // 196608 dwords = 786432 bytes

// Hardware packed f32->bf16 conversion (VOP3 v_cvt_pk_bf16_f32)
__device__ __forceinline__ unsigned int pk_bf16(float lo, float hi) {
  unsigned int r;
  asm("v_cvt_pk_bf16_f32 %0, %1, %2" : "=v"(r) : "v"(lo), "v"(hi));
  return r;
}

// Fused weight matrix Wcat[k][n], k in [0,1024), n in [0,384)
// n:   0..99  share experts (col = e*10+h) <- Ws ; 100..199 task0 <- Wt[0]
//    200..299 task1 <- Wt[1] ; 300..319 gate0 <- Wg[0]; 320..339 gate1 <- Wg[1]; pad 0
__device__ __forceinline__ float fetchW(int k, int n, const float* Ws,
                                        const float* Wt, const float* Wg) {
  if (n < 100) { int e = n / 10, h = n % 10; return Ws[(e * DK + k) * 10 + h]; }
  if (n < 300) {
    int t = (n - 100) / 100, r = (n - 100) % 100;
    int e = r / 10, h = r % 10;
    return Wt[((t * 10 + e) * DK + k) * 10 + h];
  }
  if (n < 340) { int t = (n - 300) / 20, g = (n - 300) % 20; return Wg[(t * DK + k) * 20 + g]; }
  return 0.0f;
}

// Pack Wcat into per-lane WMMA B-fragment layout:
//   Bpack[((kc*24 + t)*32 + lane)*8 + v] = bf16x2 of
//   k = kc*32 + (lane>=16 ? 16 : 0) + 2v (+1), n = t*16 + (lane&15)
__global__ void ple_prepack(const float* __restrict__ Ws, const float* __restrict__ Wt,
                            const float* __restrict__ Wg, const float* __restrict__ bs,
                            const float* __restrict__ bt, const float* __restrict__ bg,
                            unsigned int* __restrict__ Bpack, float* __restrict__ biasN) {
  int bid = blockIdx.x;          // 0..767 = kc*24 + t
  int kc = bid / NTP, t = bid % NTP;
  int tid = threadIdx.x;         // 0..255
  int lane = tid >> 3, v = tid & 7;
  int n  = t * 16 + (lane & 15);
  int k0 = kc * 32 + ((lane >= 16) ? 16 : 0) + 2 * v;
  float f0 = fetchW(k0,     n, Ws, Wt, Wg);
  float f1 = fetchW(k0 + 1, n, Ws, Wt, Wg);
  Bpack[((kc * NTP + t) * 32 + lane) * 8 + v] = pk_bf16(f0, f1);

  if (kc == 0 && v == 0 && lane < 16 && n < NCOL) {
    float b;
    if      (n < 100) b = bs[n];
    else if (n < 300) b = bt[n - 100];
    else if (n < 340) b = bg[n - 300];
    else              b = 0.0f;
    biasN[n] = b;
  }
}

// 256 threads = 8 waves = 4 row-groups (16 rows each) x 2 N-halves (11 tiles each)
__global__ __launch_bounds__(256) void ple_gemm(
    const float* __restrict__ x, const unsigned int* __restrict__ Bpack,
    const float* __restrict__ biasN, const float* __restrict__ Wc,
    const float* __restrict__ bc, const float* __restrict__ Wv,
    const float* __restrict__ bv, float* __restrict__ out) {

  __shared__ __align__(16) unsigned int sB[2][CHUNK_DW];   // 2 x 24 KB B chunks (DMA dest)
  __shared__ __align__(16) unsigned int sA[2][4 * 32 * 8]; // 2 x 4 KB A fragments
  __shared__ float sC[4][16 * NCOL];                       // per-row-group C slice (90 KB)

  const int tid   = threadIdx.x;
  const int wave  = tid >> 5, lane = tid & 31;
  const int l15   = lane & 15;
  const int waveM = wave >> 1;      // row group 0..3
  const int waveN = wave & 1;       // N half 0..1
  const int blockRow = blockIdx.x * 64;

  // Cooperative A staging map: thread j -> local row rl=j/4, 8-wide k run q=j%4
  // A-frag k order used here (contiguous, same as Bpack): lanes 0-15 hold K {0..15}; lanes 16-31 hold {16..31}
  const int rl = tid >> 2;          // 0..63
  const int q  = tid & 3;
  const float* xsrc = x + (size_t)(blockRow + rl) * DK + q * 8;
  const int saIdx = (((rl >> 4) * 32) + (rl & 15) + ((q & 2) ? 16 : 0)) * 8 + ((q & 1) ? 4 : 0);

  const unsigned sbOff =
      (unsigned)(uintptr_t)(__attribute__((address_space(3))) unsigned char*)&sB[0][0];
  const unsigned long long bbase = (unsigned long long)(uintptr_t)Bpack;

  // Issue the 6 async b128 DMAs (per thread) moving chunk kc of Bpack into sB[buf]
  auto issueB = [&](int kc, int buf) {
    const unsigned long long g0 = bbase + (unsigned long long)kc * CHUNK_BYTES;
    const unsigned          ld0 = sbOff + (unsigned)buf * CHUNK_BYTES;
#pragma unroll
    for (int it = 0; it < 6; ++it) {
      const int i = tid + it * 256;
      unsigned long long ga = g0 + (unsigned long long)i * 16u;
      unsigned           ld = ld0 + (unsigned)i * 16u;
      asm volatile("global_load_async_to_lds_b128 %0, %1, off"
                   :: "v"(ld), "v"(ga) : "memory");
    }
  };
  // Stage the 64x32 x-tile of chunk kc into sA[buf] (packed bf16, A-frag layout)
  auto stageA = [&](int kc, int buf) {
    const v4f* ap = (const v4f*)(xsrc + kc * 32);
    v4f f0 = __builtin_nontemporal_load(ap);
    v4f f1 = __builtin_nontemporal_load(ap + 1);
    uint4 pa;
    pa.x = pk_bf16(f0.x, f0.y); pa.y = pk_bf16(f0.z, f0.w);
    pa.z = pk_bf16(f1.x, f1.y); pa.w = pk_bf16(f1.z, f1.w);
    *(uint4*)&sA[buf][saIdx] = pa;
  };

  v8f acc[NTW];
#pragma unroll
  for (int t = 0; t < NTW; ++t) acc[t] = (v8f){0.f,0.f,0.f,0.f,0.f,0.f,0.f,0.f};

  // Prologue: bring chunk 0 into buffer 0
  issueB(0, 0);
  stageA(0, 0);
  asm volatile("s_wait_asynccnt 0x0" ::: "memory");
  __syncthreads();

  for (int kc = 0; kc < KC; ++kc) {
    const int cur = kc & 1, nxt = cur ^ 1;
    if (kc + 1 < KC) {        // overlap next chunk's DMA + A staging with compute
      issueB(kc + 1, nxt);
      stageA(kc + 1, nxt);
    }

    union BF { uint4 q[2]; v16bf v; };
    BF afrag;
    const uint4* apf = (const uint4*)&sA[cur][(waveM * 32 + lane) * 8];
    afrag.q[0] = apf[0]; afrag.q[1] = apf[1];

    const unsigned int* bbase_lds = &sB[cur][(waveN * NTW) * 256 + lane * 8];
    BF bcur, bnxt;
    bcur.q[0] = ((const uint4*)bbase_lds)[0];
    bcur.q[1] = ((const uint4*)bbase_lds)[1];
#pragma unroll
    for (int t = 0; t < NTW; ++t) {
      if (t + 1 < NTW) {      // prefetch next B fragment before this WMMA
        const uint4* np = (const uint4*)(bbase_lds + (t + 1) * 256);
        bnxt.q[0] = np[0]; bnxt.q[1] = np[1];
      }
      acc[t] = __builtin_amdgcn_wmma_f32_16x16x32_bf16(
          false, afrag.v, false, bcur.v, (short)0, acc[t], false, false);
      bcur = bnxt;
    }

    if (kc + 1 < KC)
      asm volatile("s_wait_asynccnt 0x0" ::: "memory");  // next chunk's DMAs landed
    __syncthreads();
  }

  // Epilogue: fused bias, transpose through LDS row-group slices
  float biasv[NTW];
#pragma unroll
  for (int t = 0; t < NTW; ++t) biasv[t] = biasN[(waveN * NTW + t) * 16 + l15];

  const int mrow = (lane >= 16) ? 8 : 0;   // C layout: vgpr r -> M = r (+8 upper lanes)
#pragma unroll
  for (int t = 0; t < NTW; ++t) {
#pragma unroll
    for (int r = 0; r < 8; ++r)
      sC[waveM][(mrow + r) * NCOL + (waveN * NTW + t) * 16 + l15] = acc[t][r] + biasv[t];
  }
  __syncthreads();   // both N-half waves contribute to each row-group slice

  if (tid < 64) {
    const float* rowp = &sC[tid >> 4][(tid & 15) * NCOL];
    const int grow = blockRow + tid;
#pragma unroll
    for (int task = 0; task < 2; ++task) {
      const float* Wh = task ? Wv : Wc;
      const float bb  = task ? bv[0] : bc[0];
      float g[20], mx = -3.0e38f;
#pragma unroll
      for (int i = 0; i < 20; ++i) { g[i] = rowp[300 + task * 20 + i]; mx = fmaxf(mx, g[i]); }
      float s = 0.f;
#pragma unroll
      for (int i = 0; i < 20; ++i) { g[i] = __expf(g[i] - mx); s += g[i]; }
      const float inv = 1.0f / s;
      float f[10];
#pragma unroll
      for (int h = 0; h < 10; ++h) f[h] = 0.f;
#pragma unroll
      for (int e = 0; e < 10; ++e) {
        const float ge = g[e] * inv, gs = g[10 + e] * inv;
#pragma unroll
        for (int h = 0; h < 10; ++h) {
          f[h] += ge * fmaxf(rowp[100 + task * 100 + e * 10 + h], 0.f)
                + gs * fmaxf(rowp[e * 10 + h], 0.f);
        }
      }
      float logit = bb;
#pragma unroll
      for (int h = 0; h < 10; ++h) logit += f[h] * Wh[h];
      out[task * BATCH + grow] = 1.0f / (1.0f + __expf(-logit));
    }
  }
}

extern "C" void kernel_launch(void* const* d_in, const int* in_sizes, int n_in,
                              void* d_out, int out_size, void* d_ws, size_t ws_size,
                              hipStream_t stream) {
  (void)in_sizes; (void)n_in; (void)out_size; (void)ws_size;
  const float* x  = (const float*)d_in[0];
  // d_in[1]=show_index (unused), d_in[2]=st (unused)
  const float* Ws = (const float*)d_in[3];
  const float* bs = (const float*)d_in[4];
  const float* Wt = (const float*)d_in[5];
  const float* bt = (const float*)d_in[6];
  const float* Wg = (const float*)d_in[7];
  const float* bg = (const float*)d_in[8];
  const float* Wc = (const float*)d_in[9];
  const float* bc = (const float*)d_in[10];
  const float* Wv = (const float*)d_in[11];
  const float* bv = (const float*)d_in[12];

  unsigned int* Bpack = (unsigned int*)d_ws;
  float* biasN = (float*)((char*)d_ws + (size_t)BPACK_DW * 4);  // +786432 B

  ple_prepack<<<KC * NTP, 256, 0, stream>>>(Ws, Wt, Wg, bs, bt, bg, Bpack, biasN);
  ple_gemm<<<BATCH / 64, 256, 0, stream>>>(x, Bpack, biasN, Wc, bc, Wv, bv,
                                           (float*)d_out);
}
